// LongformerRegressor_48936857371156
// MI455X (gfx1250) — matrix operand order repaired and checked by reading, not compiled, not run on hardware
//
#include <hip/hip_runtime.h>

typedef __attribute__((ext_vector_type(16))) __bf16 v16bf;
typedef __attribute__((ext_vector_type(8)))  __bf16 v8bf;
typedef __attribute__((ext_vector_type(8)))  float  v8f;

#define B_   4
#define S_   4096
#define D_   256
#define H_   8
#define DH_  32
#define W_   128
#define NB_  32
#define FF_  1024
#define M_   (B_*S_)          // 16384 token rows
#define NEG_ (-1.0e9f)
#define SCALE_ 0.17677669529663687f   // 1/sqrt(32)

// ---- CDNA5 async global->LDS path (guarded; falls back to sync staging) ----
#if defined(__has_builtin)
#if __has_builtin(__builtin_amdgcn_global_load_async_to_lds_b128) && \
    __has_builtin(__builtin_amdgcn_s_wait_asynccnt)
#define USE_ASYNC_LDS 1
#endif
#endif
#ifndef USE_ASYNC_LDS
#define USE_ASYNC_LDS 0
#endif

#if USE_ASYNC_LDS
// Builtin signature (from hipcc diagnostic): param0 = AS1 pointer to
// 'int __attribute__((vector_size(16)))', param1 = LDS-side pointer of same
// pointee. AS1 prints as '__device__' in HIP mode.
typedef int async_v4i __attribute__((vector_size(16)));
typedef __attribute__((address_space(1))) async_v4i ga_v4i;
typedef __attribute__((address_space(3))) async_v4i la_v4i;
// ISA 10.2: generic shared address bits [31:0] are the LDS byte offset, so an
// integer round-trip yields a valid AS3 pointer without an addrspacecast.
static __device__ __forceinline__ la_v4i* to_lds(void* p) {
  return (la_v4i*)(unsigned int)(unsigned long long)p;
}
static __device__ __forceinline__ ga_v4i* to_glb(const void* p) {
  return (ga_v4i*)(unsigned long long)p;
}
#endif

static __device__ __forceinline__ __bf16 f2bf(float f) {
  union { float f; unsigned u; } a; a.f = f;
  unsigned r = a.u + 0x7FFFu + ((a.u >> 16) & 1u);
  union { unsigned short s; __bf16 b; } o; o.s = (unsigned short)(r >> 16);
  return o.b;
}

// ---------------- fp32 -> bf16 converters ----------------
__global__ void cvt_bf16_kernel(const float* __restrict__ x, __bf16* __restrict__ y, int n) {
  int i = blockIdx.x * 256 + threadIdx.x;
  if (i < n) y[i] = f2bf(x[i]);
}
// W is K x N (row major); Wt is N x K (row major) = W transposed, bf16
__global__ void cvt_t_bf16_kernel(const float* __restrict__ w, __bf16* __restrict__ wt, int K, int N) {
  int i = blockIdx.x * 256 + threadIdx.x;
  if (i < K * N) {
    int k = i / N, n = i % N;
    wt[(size_t)n * K + k] = f2bf(w[i]);
  }
}

// ---------------- WMMA GEMM: out = A(bf16, MxK) * Bt^T + bias ----------------
// Bt is N x K bf16 (pre-transposed weights). act: 0=none, 1=exact gelu.
// Block = 128 threads (4 waves). Block tile = 64 rows x 64 cols; each wave
// computes a 16x64 strip (4 accumulators). The shared 64-col B panel is
// double-buffered in LDS via async global->LDS copies (ASYNCcnt).
__global__ __launch_bounds__(128) void wmma_gemm_kernel(
    const __bf16* __restrict__ A, const __bf16* __restrict__ Bt,
    const float* __restrict__ bias, float* __restrict__ outF,
    __bf16* __restrict__ outB, int M, int N, int K, int act)
{
  __shared__ __bf16 Bs[2][64 * 32];   // [buf][col*32 + k] : 64 cols x 32 K, 4KB each
  const int tid  = threadIdx.x;
  const int lane = tid & 31;
  const int wave = tid >> 5;
  const int n0 = blockIdx.x * 64;
  const int m0 = (blockIdx.y * 4 + wave) * 16;
  const int r  = lane & 15;
  const int hf = lane >> 4;
  const __bf16* arow = A + (size_t)(m0 + r) * K;      // A row (lane layout)

  // staging: thread covers 32 bytes (16 bf16) of one B column's K-chunk
  const int sc = tid >> 1;                            // column 0..63
  const int sh = tid & 1;                             // K half (0..15 / 16..31)
  const __bf16* gsrc = Bt + (size_t)(n0 + sc) * K + sh * 16;
  __bf16* ldst[2] = { &Bs[0][sc * 32 + sh * 16], &Bs[1][sc * 32 + sh * 16] };

  auto stage = [&](int buf, int kb) {
#if USE_ASYNC_LDS
    __builtin_amdgcn_global_load_async_to_lds_b128(to_glb(gsrc + kb), to_lds(ldst[buf]), 0, 0);
    __builtin_amdgcn_global_load_async_to_lds_b128(to_glb(gsrc + kb), to_lds(ldst[buf]), 16, 0);
#else
    *(v8bf*)(ldst[buf])     = *(const v8bf*)(gsrc + kb);
    *(v8bf*)(ldst[buf] + 8) = *(const v8bf*)(gsrc + kb + 8);
#endif
  };

  v8f acc[4] = {};
  stage(0, 0);
#if USE_ASYNC_LDS
  __builtin_amdgcn_s_wait_asynccnt(0);
#endif
  __syncthreads();

  int buf = 0;
  for (int kb = 0; kb < K; kb += 32) {
    if (kb + 32 < K) stage(buf ^ 1, kb + 32);
    __builtin_prefetch(arow + kb + 128, 0, 1);
    union { v16bf v; v8bf h[2]; } av;
    av.h[0] = *(const v8bf*)(arow + kb + hf * 16);       // K = kb+16*hf .. +7
    av.h[1] = *(const v8bf*)(arow + kb + hf * 16 + 8);   // K = kb+16*hf+8 ..
    const __bf16* bsb = Bs[buf];
#pragma unroll
    for (int nt = 0; nt < 4; ++nt) {
      v16bf bv = *(const v16bf*)&bsb[(nt * 16 + r) * 32 + hf * 16];
      acc[nt] = __builtin_amdgcn_wmma_f32_16x16x32_bf16(false, av.v, false, bv,
                                                        (short)0, acc[nt], false, false);
    }
#if USE_ASYNC_LDS
    __builtin_amdgcn_s_wait_asynccnt(0);
#endif
    __syncthreads();
    buf ^= 1;
  }

#pragma unroll
  for (int nt = 0; nt < 4; ++nt) {
    const int col = n0 + nt * 16 + r;
    const float bval = bias ? bias[col] : 0.0f;
    for (int i = 0; i < 8; ++i) {
      int row = m0 + hf * 8 + i;
      float x = acc[nt][i] + bval;
      if (act == 1) x = 0.5f * x * (1.0f + erff(x * 0.70710678118654752f));
      if (outF) outF[(size_t)row * N + col] = x;
      if (outB) outB[(size_t)row * N + col] = f2bf(x);
    }
  }
}

// ---------------- embedding + LayerNorm ----------------
__global__ __launch_bounds__(256) void embed_ln_kernel(
    const int* __restrict__ ids, const float* __restrict__ we,
    const float* __restrict__ pe, const float* __restrict__ te,
    const float* __restrict__ g, const float* __restrict__ bb,
    float* __restrict__ h, __bf16* __restrict__ hb)
{
  int tok = blockIdx.x;
  int d = threadIdx.x;
  int s = tok & (S_ - 1);
  int id = ids[tok];
  float x = we[(size_t)id * D_ + d] + pe[(size_t)(s + 2) * D_ + d] + te[d];
  __shared__ float red[256];
  red[d] = x; __syncthreads();
  for (int st = 128; st > 0; st >>= 1) { if (d < st) red[d] += red[d + st]; __syncthreads(); }
  float mean = red[0] * (1.0f / 256.0f);
  __syncthreads();
  float c = x - mean;
  red[d] = c * c; __syncthreads();
  for (int st = 128; st > 0; st >>= 1) { if (d < st) red[d] += red[d + st]; __syncthreads(); }
  float var = red[0] * (1.0f / 256.0f);
  float y = c * rsqrtf(var + 1e-12f) * g[d] + bb[d];
  h[(size_t)tok * D_ + d] = y;
  hb[(size_t)tok * D_ + d] = f2bf(y);
}

// h = LN(h + delta); also emit bf16 copy
__global__ __launch_bounds__(256) void resid_ln_kernel(
    float* __restrict__ h, const float* __restrict__ delta,
    const float* __restrict__ g, const float* __restrict__ bb, __bf16* __restrict__ hb)
{
  int tok = blockIdx.x;
  int d = threadIdx.x;
  float x = h[(size_t)tok * D_ + d] + delta[(size_t)tok * D_ + d];
  __shared__ float red[256];
  red[d] = x; __syncthreads();
  for (int st = 128; st > 0; st >>= 1) { if (d < st) red[d] += red[d + st]; __syncthreads(); }
  float mean = red[0] * (1.0f / 256.0f);
  __syncthreads();
  float c = x - mean;
  red[d] = c * c; __syncthreads();
  for (int st = 128; st > 0; st >>= 1) { if (d < st) red[d] += red[d + st]; __syncthreads(); }
  float var = red[0] * (1.0f / 256.0f);
  float y = c * rsqrtf(var + 1e-12f) * g[d] + bb[d];
  h[(size_t)tok * D_ + d] = y;
  hb[(size_t)tok * D_ + d] = f2bf(y);
}

// ---------------- sliding-window local attention (flash-style, WMMA) ----------------
// q/k/v: (B,S,D) fp32 with head hh at cols hh*32..+31. Writes attn (B,S,D).
__global__ __launch_bounds__(256) void local_attn_kernel(
    const float* __restrict__ q, const float* __restrict__ k, const float* __restrict__ v,
    const int* __restrict__ am, const int* __restrict__ gmk, float* __restrict__ attn)
{
  const int n  = blockIdx.x;   // window block 0..NB-1
  const int hh = blockIdx.y;
  const int b  = blockIdx.z;
  const int tid = threadIdx.x;

  __shared__ __bf16 Ks[384 * 32];     // key t, dim d (row major)
  __shared__ __bf16 Vt[32 * 400];     // dim d, key t (transposed, padded stride)
  __shared__ __bf16 pscr[8 * 512];    // per-wave P scratch (16x32)
  __shared__ float  sgl[128];
  __shared__ float  kgv[32], vgv[32];
  __shared__ unsigned char wok[384];

  for (int t = tid; t < 384; t += 256) {
    int sk = (n - 1) * W_ + t;
    int ok = 0;
    if (sk >= 0 && sk < S_) ok = (am[b * S_ + sk] != 0) && (gmk[b * S_ + sk] == 0);
    wok[t] = (unsigned char)ok;
  }
  for (int idx = tid; idx < 384 * 32; idx += 256) {
    int t = idx >> 5, d = idx & 31;
    int sk = (n - 1) * W_ + t;
    float kv = 0.0f, vv = 0.0f;
    if (sk >= 0 && sk < S_) {
      size_t o = ((size_t)b * S_ + sk) * D_ + hh * DH_ + d;
      kv = k[o]; vv = v[o];
    }
    Ks[t * 32 + d]  = f2bf(kv);
    Vt[d * 400 + t] = f2bf(vv);
  }
  if (tid < 32) {
    size_t o0 = ((size_t)b * S_) * D_ + hh * DH_ + tid;   // global token s=0
    kgv[tid] = k[o0];
    vgv[tid] = v[o0];
  }
  __syncthreads();
  if (tid < 128) {
    int s = n * W_ + tid;
    const float* qp = q + ((size_t)b * S_ + s) * D_ + hh * DH_;
    float acc = 0.0f;
    for (int d = 0; d < 32; ++d) acc += qp[d] * kgv[d];
    sgl[tid] = acc * SCALE_ + ((float)am[b * S_] - 1.0f) * 1.0e9f;
  }
  __syncthreads();

  const int wave = tid >> 5, lane = tid & 31;
  const int rr = lane & 15, hf = lane >> 4;
  const int rbase = wave * 16;

  // Q tile (A layout), pre-scaled
  const float* qp = q + ((size_t)b * S_ + n * W_ + rbase + rr) * D_ + hh * DH_;
  v16bf aq;
  for (int j = 0; j < 8; ++j) {
    aq[j]     = f2bf(qp[hf * 16 + j] * SCALE_);
    aq[8 + j] = f2bf(qp[hf * 16 + 8 + j] * SCALE_);
  }

  float m_[8], l_[8];
  v8f O0, O1;
  for (int i = 0; i < 8; ++i) {
    m_[i] = sgl[rbase + hf * 8 + i];   // seed with global column score
    l_[i] = 1.0f;
    O0[i] = vgv[rr];
    O1[i] = vgv[16 + rr];
  }

  __bf16* ps = pscr + wave * 512;
  v8f zf = {};
  for (int jc = 0; jc < 12; ++jc) {
    int t0 = jc * 32;
    v16bf bk0 = *(const v16bf*)&Ks[(t0 + rr) * 32 + hf * 16];
    v16bf bk1 = *(const v16bf*)&Ks[(t0 + 16 + rr) * 32 + hf * 16];
    v8f s0 = __builtin_amdgcn_wmma_f32_16x16x32_bf16(false, aq, false, bk0, (short)0, zf, false, false);
    v8f s1 = __builtin_amdgcn_wmma_f32_16x16x32_bf16(false, aq, false, bk1, (short)0, zf, false, false);

    int ta = t0 + rr, tb = ta + 16;
    int wa = wok[ta], wb = wok[tb];
    for (int i = 0; i < 8; ++i) {
      int p = rbase + hf * 8 + i;                 // q row in [0,128)
      bool oka = wa && (ta >= p) && (ta <= p + 256);
      bool okb = wb && (tb >= p) && (tb <= p + 256);
      s0[i] = oka ? s0[i] : NEG_;
      s1[i] = okb ? s1[i] : NEG_;
    }
    for (int i = 0; i < 8; ++i) {
      float rm = fmaxf(s0[i], s1[i]);
      for (int off = 1; off < 16; off <<= 1) rm = fmaxf(rm, __shfl_xor(rm, off, 32));
      float mn = fmaxf(m_[i], rm);
      float fsc = __expf(m_[i] - mn);
      m_[i] = mn;
      float p0 = __expf(s0[i] - mn), p1 = __expf(s1[i] - mn);
      float rs = p0 + p1;
      for (int off = 1; off < 16; off <<= 1) rs += __shfl_xor(rs, off, 32);
      l_[i] = l_[i] * fsc + rs;
      O0[i] *= fsc; O1[i] *= fsc;
      s0[i] = p0; s1[i] = p1;
    }
    // P (C layout) -> LDS -> A layout
    for (int i = 0; i < 8; ++i) {
      int prow = hf * 8 + i;
      ps[prow * 32 + rr]      = f2bf(s0[i]);
      ps[prow * 32 + 16 + rr] = f2bf(s1[i]);
    }
    __syncthreads();
    union { v16bf v; v8bf h[2]; } ap;
    ap.h[0] = *(const v8bf*)&ps[rr * 32 + hf * 16];
    ap.h[1] = *(const v8bf*)&ps[rr * 32 + hf * 16 + 8];
    v16bf bv0 = *(const v16bf*)&Vt[rr * 400 + t0 + hf * 16];
    v16bf bv1 = *(const v16bf*)&Vt[(16 + rr) * 400 + t0 + hf * 16];
    O0 = __builtin_amdgcn_wmma_f32_16x16x32_bf16(false, ap.v, false, bv0, (short)0, O0, false, false);
    O1 = __builtin_amdgcn_wmma_f32_16x16x32_bf16(false, ap.v, false, bv1, (short)0, O1, false, false);
    __syncthreads();
  }
  for (int i = 0; i < 8; ++i) {
    float inv = 1.0f / l_[i];
    int s = n * W_ + rbase + hf * 8 + i;
    size_t o = ((size_t)b * S_ + s) * D_ + hh * DH_;
    attn[o + rr]      = O0[i] * inv;
    attn[o + 16 + rr] = O1[i] * inv;
  }
}

// ---------------- global token projection qg = h[:,0,:] @ Wqg + bqg ----------------
__global__ __launch_bounds__(256) void qg_kernel(
    const float* __restrict__ h, const float* __restrict__ Wqg,
    const float* __restrict__ bqg, float* __restrict__ qg)
{
  int b = blockIdx.x, o = threadIdx.x;
  const float* hr = h + (size_t)b * S_ * D_;   // token 0
  float acc = bqg[o];
  for (int kk = 0; kk < D_; ++kk) acc += hr[kk] * Wqg[kk * D_ + o];
  qg[b * D_ + o] = acc;
}

// ---------------- global attention (G=1): overwrite attn row s=0 ----------------
__global__ __launch_bounds__(256) void global_attn_kernel(
    const float* __restrict__ qg, const float* __restrict__ kg, const float* __restrict__ vg,
    const int* __restrict__ am, float* __restrict__ attn)
{
  const int hh = blockIdx.x, b = blockIdx.y;
  const int tid = threadIdx.x;
  __shared__ float qgl[32];
  __shared__ float sc[S_];
  __shared__ float red[256];
  __shared__ float acc[256][33];
  if (tid < 32) qgl[tid] = qg[(size_t)b * D_ + hh * DH_ + tid] * SCALE_;
  __syncthreads();
  float lmax = -3.4e38f;
  for (int s = tid; s < S_; s += 256) {
    const float* kp = kg + ((size_t)b * S_ + s) * D_ + hh * DH_;
    float d = 0.0f;
    for (int j = 0; j < 32; ++j) d += qgl[j] * kp[j];
    d += ((float)am[b * S_ + s] - 1.0f) * 1.0e9f;
    sc[s] = d; lmax = fmaxf(lmax, d);
  }
  red[tid] = lmax; __syncthreads();
  for (int st = 128; st > 0; st >>= 1) { if (tid < st) red[tid] = fmaxf(red[tid], red[tid + st]); __syncthreads(); }
  float gmax = red[0];
  __syncthreads();
  float lsum = 0.0f, av[32];
  for (int j = 0; j < 32; ++j) av[j] = 0.0f;
  for (int s = tid; s < S_; s += 256) {
    float p = __expf(sc[s] - gmax);
    lsum += p;
    const float* vp = vg + ((size_t)b * S_ + s) * D_ + hh * DH_;
    for (int j = 0; j < 32; ++j) av[j] += p * vp[j];
  }
  red[tid] = lsum; __syncthreads();
  for (int st = 128; st > 0; st >>= 1) { if (tid < st) red[tid] += red[tid + st]; __syncthreads(); }
  float tot = red[0];
  for (int j = 0; j < 32; ++j) acc[tid][j] = av[j];
  __syncthreads();
  for (int st = 128; st > 0; st >>= 1) {
    if (tid < st) for (int j = 0; j < 32; ++j) acc[tid][j] += acc[tid + st][j];
    __syncthreads();
  }
  if (tid < 32) attn[((size_t)b * S_) * D_ + hh * DH_ + tid] = acc[0][tid] / tot;
}

// ---------------- regressor head ----------------
__global__ __launch_bounds__(512) void reg1_kernel(
    const float* __restrict__ h, const float* __restrict__ tab,
    const float* __restrict__ W, const float* __restrict__ bb, float* __restrict__ x1)
{
  int b = blockIdx.x, o = threadIdx.x;   // 512 outputs
  const float* hr = h + (size_t)b * S_ * D_;
  float acc = bb[o];
  for (int kk = 0; kk < 256; ++kk) acc += hr[kk] * W[kk * 512 + o];
  for (int kk = 0; kk < 16; ++kk)  acc += tab[b * 16 + kk] * W[(256 + kk) * 512 + o];
  x1[b * 512 + o] = fmaxf(acc, 0.0f);
}
__global__ __launch_bounds__(256) void reg2_kernel(
    const float* __restrict__ x1, const float* __restrict__ W,
    const float* __restrict__ bb, float* __restrict__ x2)
{
  int b = blockIdx.x, o = threadIdx.x;
  float acc = bb[o];
  for (int kk = 0; kk < 512; ++kk) acc += x1[b * 512 + kk] * W[kk * 256 + o];
  x2[b * 256 + o] = fmaxf(acc, 0.0f);
}
__global__ void reg3_kernel(const float* __restrict__ x2, const float* __restrict__ W,
                            const float* __restrict__ bb, float* __restrict__ out)
{
  int b = threadIdx.x;
  if (b < B_) {
    float acc = bb[0];
    for (int kk = 0; kk < 256; ++kk) acc += x2[b * 256 + kk] * W[kk];
    out[b] = acc;
  }
}

// =======================================================================
extern "C" void kernel_launch(void* const* d_in, const int* in_sizes, int n_in,
                              void* d_out, int out_size, void* d_ws, size_t ws_size,
                              hipStream_t stream) {
  (void)in_sizes; (void)n_in; (void)out_size; (void)ws_size;
  const int*   ids = (const int*)d_in[0];
  const int*   am  = (const int*)d_in[1];
  const int*   gmk = (const int*)d_in[2];
  const float* tab = (const float*)d_in[3];
  const float* we  = (const float*)d_in[4];
  const float* pe  = (const float*)d_in[5];
  const float* te  = (const float*)d_in[6];
  const float* es  = (const float*)d_in[7];
  const float* eb  = (const float*)d_in[8];
  const float* Wq  = (const float*)d_in[9];
  const float* Wk  = (const float*)d_in[10];
  const float* Wv  = (const float*)d_in[11];
  const float* Wqg = (const float*)d_in[12];
  const float* Wkg = (const float*)d_in[13];
  const float* Wvg = (const float*)d_in[14];
  const float* Wo  = (const float*)d_in[15];
  const float* Wi  = (const float*)d_in[16];
  const float* Wf  = (const float*)d_in[17];
  const float* bq  = (const float*)d_in[18];
  const float* bk  = (const float*)d_in[19];
  const float* bv  = (const float*)d_in[20];
  const float* bqg = (const float*)d_in[21];
  const float* bkg = (const float*)d_in[22];
  const float* bvg = (const float*)d_in[23];
  const float* bo  = (const float*)d_in[24];
  const float* bi  = (const float*)d_in[25];
  const float* bfw = (const float*)d_in[26];
  const float* l1s = (const float*)d_in[27];
  const float* l1b = (const float*)d_in[28];
  const float* l2s = (const float*)d_in[29];
  const float* l2b = (const float*)d_in[30];
  const float* Wr1 = (const float*)d_in[31];
  const float* br1 = (const float*)d_in[32];
  const float* Wr2 = (const float*)d_in[33];
  const float* br2 = (const float*)d_in[34];
  const float* Wr3 = (const float*)d_in[35];
  const float* br3 = (const float*)d_in[36];
  float* outp = (float*)d_out;

  // ---- workspace carve ----
  char* base = (char*)d_ws;
  size_t off = 0;
  auto carve = [&](size_t bytes) -> void* {
    void* p = base + off;
    off += (bytes + 255) & ~(size_t)255;
    return p;
  };
  float*  h    = (float*) carve((size_t)M_ * D_ * 4);
  __bf16* hb   = (__bf16*)carve((size_t)M_ * D_ * 2);
  float*  qb   = (float*) carve((size_t)M_ * D_ * 4);
  float*  kb2  = (float*) carve((size_t)M_ * D_ * 4);
  float*  vb   = (float*) carve((size_t)M_ * D_ * 4);
  float*  attn = (float*) carve((size_t)M_ * D_ * 4);
  __bf16* bigb = (__bf16*)carve((size_t)M_ * FF_ * 2);
  const size_t perL = 6 * 65536 + 2 * 262144;            // bf16 elems per layer
  __bf16* wts  = (__bf16*)carve((size_t)4 * perL * 2);
  float*  qg   = (float*) carve((size_t)B_ * D_ * 4);
  float*  x1   = (float*) carve((size_t)B_ * 512 * 4);
  float*  x2   = (float*) carve((size_t)B_ * 256 * 4);

  // ---- one-time (per launch) weight convert+transpose to bf16 ----
  const int gDD = (65536 + 255) / 256, gDF = (262144 + 255) / 256;
  for (int l = 0; l < 4; ++l) {
    __bf16* wl = wts + (size_t)l * perL;
    cvt_t_bf16_kernel<<<gDD, 256, 0, stream>>>(Wq  + (size_t)l * 65536,  wl + 0 * 65536, D_, D_);
    cvt_t_bf16_kernel<<<gDD, 256, 0, stream>>>(Wk  + (size_t)l * 65536,  wl + 1 * 65536, D_, D_);
    cvt_t_bf16_kernel<<<gDD, 256, 0, stream>>>(Wv  + (size_t)l * 65536,  wl + 2 * 65536, D_, D_);
    cvt_t_bf16_kernel<<<gDD, 256, 0, stream>>>(Wkg + (size_t)l * 65536,  wl + 3 * 65536, D_, D_);
    cvt_t_bf16_kernel<<<gDD, 256, 0, stream>>>(Wvg + (size_t)l * 65536,  wl + 4 * 65536, D_, D_);
    cvt_t_bf16_kernel<<<gDD, 256, 0, stream>>>(Wo  + (size_t)l * 65536,  wl + 5 * 65536, D_, D_);
    cvt_t_bf16_kernel<<<gDF, 256, 0, stream>>>(Wi  + (size_t)l * 262144, wl + 6 * 65536, D_, FF_);
    cvt_t_bf16_kernel<<<gDF, 256, 0, stream>>>(Wf  + (size_t)l * 262144, wl + 6 * 65536 + 262144, FF_, D_);
  }

  auto gemm = [&](const __bf16* A, const __bf16* Bt, const float* bias,
                  float* oF, __bf16* oB, int N, int K, int act) {
    dim3 grid(N / 64, M_ / 64);
    wmma_gemm_kernel<<<grid, 128, 0, stream>>>(A, Bt, bias, oF, oB, M_, N, K, act);
  };

  // ---- embeddings + LN ----
  embed_ln_kernel<<<M_, 256, 0, stream>>>(ids, we, pe, te, es, eb, h, hb);

  for (int l = 0; l < 4; ++l) {
    __bf16* wl = wts + (size_t)l * perL;
    // q/k/v projections
    gemm(hb, wl + 0 * 65536, bq + l * D_, qb,  nullptr, D_, D_, 0);
    gemm(hb, wl + 1 * 65536, bk + l * D_, kb2, nullptr, D_, D_, 0);
    gemm(hb, wl + 2 * 65536, bv + l * D_, vb,  nullptr, D_, D_, 0);
    // sliding-window attention (all rows)
    local_attn_kernel<<<dim3(NB_, H_, B_), 256, 0, stream>>>(qb, kb2, vb, am, gmk, attn);
    // global attention: kg -> qb, vg -> kb2 (q/k/v no longer needed)
    gemm(hb, wl + 3 * 65536, bkg + l * D_, qb,  nullptr, D_, D_, 0);
    gemm(hb, wl + 4 * 65536, bvg + l * D_, kb2, nullptr, D_, D_, 0);
    qg_kernel<<<B_, 256, 0, stream>>>(h, Wqg + (size_t)l * 65536, bqg + l * D_, qg);
    global_attn_kernel<<<dim3(H_, B_), 256, 0, stream>>>(qg, qb, kb2, am, attn);
    // output projection + residual LN1
    cvt_bf16_kernel<<<(M_ * D_ + 255) / 256, 256, 0, stream>>>(attn, bigb, M_ * D_);
    gemm(bigb, wl + 5 * 65536, bo + l * D_, vb, nullptr, D_, D_, 0);
    resid_ln_kernel<<<M_, 256, 0, stream>>>(h, vb, l1s + l * D_, l1b + l * D_, hb);
    // FFN: gelu(h@Wi+bi) (bf16) @ Wf + bf, residual LN2
    gemm(hb, wl + 6 * 65536, bi + l * FF_, nullptr, bigb, FF_, D_, 1);
    gemm(bigb, wl + 6 * 65536 + 262144, bfw + l * D_, vb, nullptr, D_, FF_, 0);
    resid_ln_kernel<<<M_, 256, 0, stream>>>(h, vb, l2s + l * D_, l2b + l * D_, hb);
  }

  // ---- regressor head ----
  reg1_kernel<<<B_, 512, 0, stream>>>(h, tab, Wr1, br1, x1);
  reg2_kernel<<<B_, 256, 0, stream>>>(x1, Wr2, br2, x2);
  reg3_kernel<<<1, 32, 0, stream>>>(x2, Wr3, br3, outp);
}
